// GGNN_89618787598506
// MI455X (gfx1250) — matrix operand order repaired and checked by reading, not compiled, not run on hardware
//
#include <hip/hip_runtime.h>
#include <hip/hip_bf16.h>
#include <math.h>

// Problem constants (from reference)
#define GB 8
#define GN 10000
#define GE 160000
#define GD 128
#define GL 4

#define KS (GD / 32)   // 4 K-steps of 32 per 128-deep GEMM
#define TN (GD / 16)   // 8 column tiles

typedef __attribute__((ext_vector_type(16))) __bf16 v16bf;
typedef __attribute__((ext_vector_type(8)))  float  v8f;

#define WMMA_BF16(a, b, c) \
  __builtin_amdgcn_wmma_f32_16x16x32_bf16(false, (a), false, (b), (short)0, (c), false, false)

__device__ __forceinline__ float sigmoidf_(float x) { return 1.0f / (1.0f + __expf(-x)); }

// A fragment (16x32 bf16) from a bf16 row-major matrix.
// Contiguous-k layout (matches packed B): lanes 0-15 hold M=lane, K = 0..15; lanes 16-31 hold M=lane-16,
// K = 16..31. Two contiguous 16-byte loads per lane.
__device__ __forceinline__ v16bf load_a_bf16p(const __bf16* __restrict__ A, int row, int ld,
                                              int k0, int lane) {
  const int kh = (lane & 16) ? 16 : 0;
  const __bf16* rp = A + (size_t)row * ld + k0;
  v16bf f;
#pragma unroll
  for (int i = 0; i < 8; ++i) f[i] = rp[kh + i];
#pragma unroll
  for (int i = 0; i < 8; ++i) f[8 + i] = rp[kh + 8 + i];
  return f;
}

// Packed B fragment: 16 bf16 per lane, contiguous 32 bytes, laid out at pack time
// in the contiguous-k lane layout of load_a_bf16p (lanes 0-15: N=lane,K=0..15; 16-31: K=16..31).
__device__ __forceinline__ v16bf load_b_packed(const __bf16* __restrict__ P, int fragIdx,
                                               int lane) {
  return *(const v16bf*)(P + ((size_t)fragIdx * 32 + lane) * 16);
}

// ---------------- Packing kernels (run once per launch; weights are tiny) --
// Row-major B source: B(k,n) = W[m*D*D + k*D + n]  (ggnn per-layer weights)
__global__ void pack_b_kernel(const float* __restrict__ W, __bf16* __restrict__ P,
                              long long total) {
  long long idx = (long long)blockIdx.x * blockDim.x + threadIdx.x;
  if (idx >= total) return;
  int i = (int)(idx & 15);
  int lane = (int)((idx >> 4) & 31);
  long long r = idx >> 9;                 // fragIdx = (m*TN + tn)*KS + ks
  int ks = (int)(r % KS); r /= KS;
  int tn = (int)(r % TN);
  int m  = (int)(r / TN);
  int col = tn * 16 + (lane & 15);
  int kh  = (lane & 16) ? 16 : 0;
  int k   = ks * 32 + kh + i;
  P[idx] = (__bf16)W[(size_t)m * GD * GD + (size_t)k * GD + col];
}

// Transposed B source: B(k,n) = W[(sec*D + n)*D + k]   (agg @ W_ih^T style)
__global__ void pack_bt_kernel(const float* __restrict__ W, __bf16* __restrict__ P,
                               long long total) {
  long long idx = (long long)blockIdx.x * blockDim.x + threadIdx.x;
  if (idx >= total) return;
  int i = (int)(idx & 15);
  int lane = (int)((idx >> 4) & 31);
  long long r = idx >> 9;                 // fragIdx = (sec*TN + tn)*KS + ks
  int ks = (int)(r % KS); r /= KS;
  int tn = (int)(r % TN);
  int sec = (int)(r / TN);
  int col = tn * 16 + (lane & 15);
  int kh  = (lane & 16) ? 16 : 0;
  int k   = ks * 32 + kh + i;
  P[idx] = (__bf16)W[((size_t)sec * GD + col) * GD + k];
}

// fp32 -> bf16 row-major staging pass (h / agg, once per layer)
__global__ void tobf16_kernel(const float* __restrict__ src, __bf16* __restrict__ dst,
                              long long total) {
  long long idx = (long long)blockIdx.x * blockDim.x + threadIdx.x;
  if (idx < total) dst[idx] = (__bf16)src[idx];
}

// ---------------- embedding gather -----------------------------------------
__global__ void embed_kernel(const int* __restrict__ x, const float* __restrict__ ew,
                             float* __restrict__ h, long long total) {
  long long idx = (long long)blockIdx.x * blockDim.x + threadIdx.x;
  if (idx >= total) return;
  int d = (int)(idx & (GD - 1));
  long long node = idx >> 7;
  h[idx] = ew[(size_t)x[node] * GD + d];
}

// ---------------- m = h @ W_l (bf16 WMMA; 16x128 strip per wave) -----------
// One wave computes a full 16-row x 128-col strip: each A fragment is loaded once
// and reused by 8 WMMAs (one per packed B column tile) -> 32 WMMAs, 72 b128 loads.
__global__ __launch_bounds__(32) void gemm_layer_kernel(const __bf16* __restrict__ Hb,
                                                        const __bf16* __restrict__ Wp,
                                                        float* __restrict__ Mout) {
  const int tm = blockIdx.x, b = blockIdx.y;
  const int lane = threadIdx.x;
  const __bf16* Hg = Hb + (size_t)b * GN * GD;
  float* Mb = Mout + (size_t)b * GN * GD;
  const int row = tm * 16 + (lane & 15);

  v8f acc[TN];
#pragma unroll
  for (int t = 0; t < TN; ++t) acc[t] = (v8f){};

#pragma unroll
  for (int ks = 0; ks < KS; ++ks) {
    v16bf a = load_a_bf16p(Hg, row, GD, ks * 32, lane);
#pragma unroll
    for (int t = 0; t < TN; ++t) {
      v16bf bb = load_b_packed(Wp, t * KS + ks, lane);
      acc[t] = WMMA_BF16(a, bb, acc[t]);
    }
  }
  const int rbase = tm * 16 + ((lane & 16) ? 8 : 0);
  const int cbase = lane & 15;
#pragma unroll
  for (int t = 0; t < TN; ++t)
#pragma unroll
    for (int v = 0; v < 8; ++v)
      Mb[(size_t)(rbase + v) * GD + t * 16 + cbase] = acc[t][v];
}

// ---------------- zero agg --------------------------------------------------
__global__ void zero_kernel(float* __restrict__ p, long long total) {
  long long idx = (long long)blockIdx.x * blockDim.x + threadIdx.x;
  if (idx < total) p[idx] = 0.0f;
}

// ---------------- agg[dst] += m[src] (native fp32 atomics) -----------------
__global__ void edge_scatter_kernel(const int* __restrict__ ei, const float* __restrict__ Mm,
                                    float* __restrict__ agg, long long total) {
  long long idx = (long long)blockIdx.x * blockDim.x + threadIdx.x;
  if (idx >= total) return;
  const int chunks = GD / 4;
  const long long per_graph = (long long)GE * chunks;
  int b = (int)(idx / per_graph);
  long long rem = idx - (long long)b * per_graph;
  int e = (int)(rem / chunks);
  int c = (int)(rem - (long long)e * chunks);
  int src = ei[(size_t)b * 2 * GE + e];
  int dst = ei[(size_t)b * 2 * GE + GE + e];
  const float4 v = *(const float4*)(Mm + ((size_t)b * GN + src) * GD + c * 4);
  float* ap = agg + ((size_t)b * GN + dst) * GD + c * 4;
  unsafeAtomicAdd(ap + 0, v.x);
  unsafeAtomicAdd(ap + 1, v.y);
  unsafeAtomicAdd(ap + 2, v.z);
  unsafeAtomicAdd(ap + 3, v.w);
}

// ---------------- fused GRU cell (24 WMMAs / 16x16 h' tile; packed weights) -
__global__ __launch_bounds__(32) void gru_kernel(const __bf16* __restrict__ Aggb,
                                                 const __bf16* __restrict__ Hb,
                                                 const float* __restrict__ Hf,
                                                 const __bf16* __restrict__ Pih,
                                                 const __bf16* __restrict__ Phh,
                                                 const float* __restrict__ b_ih,
                                                 const float* __restrict__ b_hh,
                                                 float* __restrict__ Hnew) {
  const int tm = blockIdx.x, tn = blockIdx.y, b = blockIdx.z;
  const int lane = threadIdx.x;
  const __bf16* Ag = Aggb + (size_t)b * GN * GD;
  const __bf16* Hg = Hb + (size_t)b * GN * GD;
  const float* Hfg = Hf + (size_t)b * GN * GD;
  float* Ob = Hnew + (size_t)b * GN * GD;
  const int row = tm * 16 + (lane & 15);
  const int col = tn * 16 + (lane & 15);

  v8f ir = {}, iz = {}, in_ = {}, hr = {}, hz = {}, hn = {};
#pragma unroll
  for (int ks = 0; ks < KS; ++ks) {
    v16bf Aa = load_a_bf16p(Ag, row, GD, ks * 32, lane);
    v16bf Ah = load_a_bf16p(Hg, row, GD, ks * 32, lane);
    v16bf Bir = load_b_packed(Pih, (0 * TN + tn) * KS + ks, lane);
    v16bf Biz = load_b_packed(Pih, (1 * TN + tn) * KS + ks, lane);
    v16bf Bin = load_b_packed(Pih, (2 * TN + tn) * KS + ks, lane);
    v16bf Bhr = load_b_packed(Phh, (0 * TN + tn) * KS + ks, lane);
    v16bf Bhz = load_b_packed(Phh, (1 * TN + tn) * KS + ks, lane);
    v16bf Bhn = load_b_packed(Phh, (2 * TN + tn) * KS + ks, lane);
    ir  = WMMA_BF16(Aa, Bir, ir);
    iz  = WMMA_BF16(Aa, Biz, iz);
    in_ = WMMA_BF16(Aa, Bin, in_);
    hr  = WMMA_BF16(Ah, Bhr, hr);
    hz  = WMMA_BF16(Ah, Bhz, hz);
    hn  = WMMA_BF16(Ah, Bhn, hn);
  }
  const float bir = b_ih[col], biz = b_ih[col + GD], bin = b_ih[col + 2 * GD];
  const float bhr = b_hh[col], bhz = b_hh[col + GD], bhn = b_hh[col + 2 * GD];
  const int rbase = tm * 16 + ((lane & 16) ? 8 : 0);
#pragma unroll
  for (int v = 0; v < 8; ++v) {
    const size_t off = (size_t)(rbase + v) * GD + col;
    float hv = Hfg[off];                      // exact fp32 h for the z*h term
    float r = sigmoidf_((ir[v] + bir) + (hr[v] + bhr));
    float z = sigmoidf_((iz[v] + biz) + (hz[v] + bhz));
    float n = tanhf((in_[v] + bin) + r * (hn[v] + bhn));
    Ob[off] = (1.0f - z) * n + z * hv;
  }
}

// ---------------- attention readout ----------------------------------------
__global__ void gate_kernel(const float* __restrict__ h, const float* __restrict__ gw,
                            const float* __restrict__ gb, float* __restrict__ eout,
                            long long totalNodes) {
  long long idx = (long long)blockIdx.x * blockDim.x + threadIdx.x;
  if (idx >= totalNodes) return;
  const float* hp = h + (size_t)idx * GD;
  float acc = 0.0f;
#pragma unroll 4
  for (int d = 0; d < GD; ++d) acc += hp[d] * gw[d];
  eout[idx] = __expf(sigmoidf_(acc + gb[0]));  // sigmoid in (0,1): exp is safe
}

__global__ __launch_bounds__(128) void attn_reduce_kernel(const float* __restrict__ h,
                                                          const float* __restrict__ e,
                                                          float* __restrict__ out) {
  const int b = blockIdx.x;
  const int d = threadIdx.x;
  const float* hb = h + (size_t)b * GN * GD;
  const float* eb = e + (size_t)b * GN;
  float acc = 0.0f, es = 0.0f;
  for (int n = 0; n < GN; ++n) {
    float en = eb[n];
    es += en;
    acc += en * hb[(size_t)n * GD + d];
  }
  out[(size_t)b * GD + d] = acc / es;
}

static inline unsigned cdiv_ll(long long a, long long b) { return (unsigned)((a + b - 1) / b); }

extern "C" void kernel_launch(void* const* d_in, const int* in_sizes, int n_in,
                              void* d_out, int out_size, void* d_ws, size_t ws_size,
                              hipStream_t stream) {
  const int*   x       = (const int*)d_in[0];
  const int*   ei      = (const int*)d_in[1];
  const float* embed_w = (const float*)d_in[2];
  const float* ggnn_w  = (const float*)d_in[3];
  const float* W_ih    = (const float*)d_in[4];
  const float* W_hh    = (const float*)d_in[5];
  const float* b_ih    = (const float*)d_in[6];
  const float* b_hh    = (const float*)d_in[7];
  const float* gate_w  = (const float*)d_in[8];
  const float* gate_b  = (const float*)d_in[9];
  float* out = (float*)d_out;

  const long long BND = (long long)GB * GN * GD;        // 10.24M floats
  const long long PGG = (long long)GL * TN * KS * 512;  // 65536 bf16 (packed ggnn)
  const long long PT  = (long long)3 * TN * KS * 512;   // 49152 bf16 (packed ih/hh)

  float* ws = (float*)d_ws;
  long long off = 0;
  float* h    = ws + off; off += BND;                // fp32 h
  float* tmp  = ws + off; off += BND;                // m / h_new
  float* agg  = ws + off; off += BND;                // fp32 agg
  float* eg   = ws + off; off += (long long)GB * GN; // gate exps
  __bf16* hb   = (__bf16*)(ws + off); off += BND / 2;   // bf16 h stage
  __bf16* aggb = (__bf16*)(ws + off); off += BND / 2;   // bf16 agg stage
  __bf16* pgg  = (__bf16*)(ws + off); off += PGG / 2;   // packed ggnn weights
  __bf16* pih  = (__bf16*)(ws + off); off += PT / 2;    // packed W_ih (B^T layout)
  __bf16* phh  = (__bf16*)(ws + off);                   // packed W_hh

  // Pack weights into WMMA B-fragment layout (once; tiny)
  pack_b_kernel<<<cdiv_ll(PGG, 256), 256, 0, stream>>>(ggnn_w, pgg, PGG);
  pack_bt_kernel<<<cdiv_ll(PT, 256), 256, 0, stream>>>(W_ih, pih, PT);
  pack_bt_kernel<<<cdiv_ll(PT, 256), 256, 0, stream>>>(W_hh, phh, PT);

  // Embedding lookup
  embed_kernel<<<cdiv_ll(BND, 256), 256, 0, stream>>>(x, embed_w, h, BND);

  const dim3 strips(GN / 16, GB);                    // 625 x 8, one wave per 16x128 strip
  const dim3 tiles(GN / 16, GD / 16, GB);            // 625 x 8 x 8, one wave per 16x16 tile
  const long long scatter_total = (long long)GB * GE * (GD / 4);

  for (int l = 0; l < GL; ++l) {
    // stage h in bf16 once per layer (feeds both GEMM and GRU A-operands)
    tobf16_kernel<<<cdiv_ll(BND, 256), 256, 0, stream>>>(h, hb, BND);
    // m = h @ W_l
    gemm_layer_kernel<<<strips, 32, 0, stream>>>(hb, pgg + (size_t)l * TN * KS * 512, tmp);
    // agg = 0 ; agg[dst] += m[src]
    zero_kernel<<<cdiv_ll(BND, 256), 256, 0, stream>>>(agg, BND);
    edge_scatter_kernel<<<cdiv_ll(scatter_total, 256), 256, 0, stream>>>(ei, tmp, agg,
                                                                         scatter_total);
    // stage agg in bf16, then fused GRU -> tmp
    tobf16_kernel<<<cdiv_ll(BND, 256), 256, 0, stream>>>(agg, aggb, BND);
    gru_kernel<<<tiles, 32, 0, stream>>>(aggb, hb, h, pih, phh, b_ih, b_hh, tmp);
    float* t = h; h = tmp; tmp = t;
  }

  // Global attention readout
  gate_kernel<<<cdiv_ll((long long)GB * GN, 256), 256, 0, stream>>>(h, gate_w, gate_b, eg,
                                                                    (long long)GB * GN);
  attn_reduce_kernel<<<GB, 128, 0, stream>>>(h, eg, out);
}
